// SelfAttention_15710990369177
// MI455X (gfx1250) — matrix operand order repaired and checked by reading, not compiled, not run on hardware
//
#include <hip/hip_runtime.h>
#include <hip/hip_bf16.h>

// ---------------------------------------------------------------------------
// Self-attention, S=8192, d_in=1024, d_out=128, fp32 in/out.
// Stage 1: QKV projection via v_wmma_f32_16x16x32_f16 (fp32->f16 on the fly).
//          Q is pre-scaled by 1/sqrt(128); K stored transposed (Kt[d][j]) so
//          QK^T B-fragments are contiguous 32B loads (packed v8h stores).
// Stage 2: flash attention (online softmax), one wave per 16-row query tile.
//          S-tile and O += P*V both via WMMA; P re-layout D->A through LDS;
//          next KV block prefetched with global_prefetch_b8.
// ---------------------------------------------------------------------------

typedef _Float16 half_t;
typedef __attribute__((ext_vector_type(16))) _Float16 v16h;
typedef __attribute__((ext_vector_type(8)))  _Float16 v8h;
typedef __attribute__((ext_vector_type(8)))  float    v8f;

#define SEQ   8192
#define D_IN  1024
#define D_OUT 128
#define QK_SCALE 0.08838834764831845f   // 1/sqrt(128)

static __device__ __forceinline__ v8f wmma_f16(v16h a, v16h b, v8f c) {
  return __builtin_amdgcn_wmma_f32_16x16x32_f16(
      /*neg_a=*/false, a, /*neg_b=*/false, b,
      /*c_mod=*/(short)0, c, /*reuse_a=*/false, /*reuse_b=*/false);
}

// ---------------------------------------------------------------------------
// Kernel 1: fused QKV projection.
// grid = (SEQ/16, 6), block = 128 (4 waves). Wave w of block (bx, by) computes
// output tile (row tile bx, column tile by*4+w) where column tiles 0..7 -> Q,
// 8..15 -> K (stored transposed), 16..23 -> V.
// ---------------------------------------------------------------------------
__global__ __launch_bounds__(128) void qkv_proj_kernel(
    const float* __restrict__ x,
    const float* __restrict__ wq,
    const float* __restrict__ wk,
    const float* __restrict__ wv,
    half_t* __restrict__ Qh,   // [SEQ][D_OUT], pre-scaled by 1/sqrt(128)
    half_t* __restrict__ Kt,   // [D_OUT][SEQ]  (transposed K)
    half_t* __restrict__ Vh)   // [SEQ][D_OUT]
{
  const int wave = threadIdx.x >> 5;
  const int lane = threadIdx.x & 31;
  const int g = lane >> 4;       // lane-half (0/1)
  const int n = lane & 15;

  const int colTileG = blockIdx.y * 4 + wave;  // 0..23
  const int mat = colTileG >> 3;               // 0=Q 1=K 2=V
  const int ct  = colTileG & 7;                // column tile within 128
  const int rowBase = blockIdx.x * 16;

  const float* __restrict__ W = (mat == 0) ? wq : ((mat == 1) ? wk : wv);

  v8f acc = {};

  const int rowA = rowBase + n;                // A-layout: M = lane%16
  const float* __restrict__ xr0 = x + (size_t)rowA * D_IN;

  for (int kk = 0; kk < D_IN; kk += 32) {
    // ---- A fragment (16x32 f16): lane-half g covers K = g*8 + {0..7, 16..23}
    const float* xr = xr0 + kk + g * 8;
    const float4 f0 = *(const float4*)(xr + 0);
    const float4 f1 = *(const float4*)(xr + 4);
    const float4 f2 = *(const float4*)(xr + 16);
    const float4 f3 = *(const float4*)(xr + 20);
    v16h a;
    a[0] = (half_t)f0.x;  a[1] = (half_t)f0.y;  a[2] = (half_t)f0.z;  a[3] = (half_t)f0.w;
    a[4] = (half_t)f1.x;  a[5] = (half_t)f1.y;  a[6] = (half_t)f1.z;  a[7] = (half_t)f1.w;
    a[8] = (half_t)f2.x;  a[9] = (half_t)f2.y;  a[10]= (half_t)f2.z;  a[11]= (half_t)f2.w;
    a[12]= (half_t)f3.x;  a[13]= (half_t)f3.y;  a[14]= (half_t)f3.z;  a[15]= (half_t)f3.w;

    // ---- B fragment (32x16 f16): lane = N column, K = g*8 + {0..7, 16..23}
    const float* wr = W + (size_t)(kk + g * 8) * D_OUT + ct * 16 + n;
    v16h b;
#pragma unroll
    for (int i = 0; i < 8; ++i) {
      b[i] = (half_t)wr[(size_t)i * D_OUT];
      b[8 + i] = (half_t)wr[(size_t)(16 + i) * D_OUT];
    }

    acc = wmma_f16(a, b, acc);
  }

  // ---- store C/D: lane holds (M = g*8 + r, N = ct*16 + n), r = 0..7.
  //      Wave-uniform branch (no per-element exec juggling).
  const int colBase = ct * 16;
  if (mat == 1) {
    // Kt[col][row]: consecutive r -> consecutive rows -> contiguous halves.
    v8h pack;
#pragma unroll
    for (int r = 0; r < 8; ++r) pack[r] = (half_t)acc[r];
    *(v8h*)(Kt + (size_t)(colBase + n) * SEQ + rowBase + g * 8) = pack;
  } else if (mat == 0) {
#pragma unroll
    for (int r = 0; r < 8; ++r) {
      const int row = rowBase + g * 8 + r;
      Qh[(size_t)row * D_OUT + colBase + n] = (half_t)(acc[r] * QK_SCALE);
    }
  } else {
#pragma unroll
    for (int r = 0; r < 8; ++r) {
      const int row = rowBase + g * 8 + r;
      Vh[(size_t)row * D_OUT + colBase + n] = (half_t)acc[r];
    }
  }
}

// ---------------------------------------------------------------------------
// Kernel 2: flash attention, one wave (32 threads) per 16-query-row tile.
// grid = SEQ/16 = 512, block = 32.
// ---------------------------------------------------------------------------
#define PT_STRIDE 48  // halves per LDS row: 32 data + 16 pad (keeps 16B align)

__global__ __launch_bounds__(32) void flash_attn_kernel(
    const half_t* __restrict__ Qh,   // [SEQ][D_OUT], pre-scaled
    const half_t* __restrict__ Kt,   // [D_OUT][SEQ]
    const half_t* __restrict__ Vh,   // [SEQ][D_OUT]
    float* __restrict__ out)         // [SEQ][D_OUT] fp32
{
  __shared__ half_t pT[16 * PT_STRIDE];   // P-tile staging, 1.5 KB

  const int lane = threadIdx.x & 31;
  const int g = lane >> 4;
  const int n = lane & 15;
  const int rowBase = blockIdx.x * 16;

  // ---- load Q tile (16x128) as 4 A-fragments (each 16x32)
  v16h qf[4];
#pragma unroll
  for (int t = 0; t < 4; ++t) {
    const half_t* qr = Qh + (size_t)(rowBase + n) * D_OUT + t * 32 + g * 8;
    const v8h lo = *(const v8h*)(qr);        // K = t*32 + g*8 + 0..7
    const v8h hi = *(const v8h*)(qr + 16);   // K = t*32 + 16 + g*8 + 0..7
#pragma unroll
    for (int i = 0; i < 8; ++i) { qf[t][i] = lo[i]; qf[t][8 + i] = hi[i]; }
  }

  v8f o[8];
#pragma unroll
  for (int c = 0; c < 8; ++c) o[c] = (v8f){};
  float mrow[8], lrow[8];
#pragma unroll
  for (int r = 0; r < 8; ++r) { mrow[r] = -INFINITY; lrow[r] = 0.0f; }

  for (int jb = 0; jb < SEQ; jb += 32) {
    // ---- prefetch next KV block (global_prefetch_b8) while WMMAs run
    if (jb + 32 < SEQ) {
#pragma unroll
      for (int t = 0; t < 4; ++t)
        __builtin_prefetch(Kt + (size_t)(t * 32 + lane) * SEQ + jb + 32, 0, 3);
      __builtin_prefetch(Vh + (size_t)(jb + 32 + lane) * D_OUT, 0, 3);
    }

    // ---- S = Q K^T for 32 key columns: two 16x16 f32 tiles
    v8f s0 = {}, s1 = {};
#pragma unroll
    for (int t = 0; t < 4; ++t) {
      const half_t* kb = Kt + (size_t)(t * 32 + g * 8) * SEQ + jb + n;   // B: lane = j col
      v16h b0, b1;
#pragma unroll
      for (int i = 0; i < 8; ++i) {
        b0[i] = kb[(size_t)i * SEQ];        b0[8 + i] = kb[(size_t)(16 + i) * SEQ];
        b1[i] = kb[(size_t)i * SEQ + 16];   b1[8 + i] = kb[(size_t)(16 + i) * SEQ + 16];
      }
      s0 = wmma_f16(qf[t], b0, s0);
      s1 = wmma_f16(qf[t], b1, s1);
    }

    // ---- online softmax (row stats live in lane-halves; xor 1/2/4/8 reduces
    //      across the 16 lanes that share a row group). Scale pre-folded in Q.
    float alpha[8];
#pragma unroll
    for (int r = 0; r < 8; ++r) {
      const float a0 = s0[r];
      const float a1 = s1[r];
      float mx = fmaxf(a0, a1);
      mx = fmaxf(mx, __shfl_xor(mx, 1, 32));
      mx = fmaxf(mx, __shfl_xor(mx, 2, 32));
      mx = fmaxf(mx, __shfl_xor(mx, 4, 32));
      mx = fmaxf(mx, __shfl_xor(mx, 8, 32));
      const float mnew = fmaxf(mrow[r], mx);
      const float al = __expf(mrow[r] - mnew);     // 0 on first block
      const float p0 = __expf(a0 - mnew);
      const float p1 = __expf(a1 - mnew);
      float ps = p0 + p1;
      ps += __shfl_xor(ps, 1, 32);
      ps += __shfl_xor(ps, 2, 32);
      ps += __shfl_xor(ps, 4, 32);
      ps += __shfl_xor(ps, 8, 32);
      lrow[r] = lrow[r] * al + ps;
      mrow[r] = mnew;
      alpha[r] = al;
      // stage P (D-layout -> LDS row-major [16][32])
      pT[(g * 8 + r) * PT_STRIDE + n]      = (half_t)p0;
      pT[(g * 8 + r) * PT_STRIDE + 16 + n] = (half_t)p1;
    }

    // wave-internal LDS visibility (lanes read other lanes' stores; LDS ops
    // are in-order per wave, so next-iteration WAR is safe)
    asm volatile("s_wait_dscnt 0" ::: "memory");

    // ---- reload P as A-fragment (16x32): lane row = n, K runs g*8 based
    const half_t* pr = pT + n * PT_STRIDE + g * 8;
    const v8h plo = *(const v8h*)(pr);        // K = g*8 + 0..7
    const v8h phi = *(const v8h*)(pr + 16);   // K = 16 + g*8 + 0..7
    v16h pa;
#pragma unroll
    for (int i = 0; i < 8; ++i) { pa[i] = plo[i]; pa[8 + i] = phi[i]; }

    // ---- rescale O and accumulate O += P * V
#pragma unroll
    for (int c = 0; c < 8; ++c) {
#pragma unroll
      for (int r = 0; r < 8; ++r) o[c][r] *= alpha[r];
    }
#pragma unroll
    for (int c = 0; c < 8; ++c) {
      const half_t* vb = Vh + (size_t)(jb + g * 8) * D_OUT + c * 16 + n;
      v16h bv;
#pragma unroll
      for (int i = 0; i < 8; ++i) { bv[i] = vb[(size_t)i * D_OUT]; bv[8 + i] = vb[(size_t)(16 + i) * D_OUT]; }
      o[c] = wmma_f16(pa, bv, o[c]);
    }
  }

  // ---- epilogue: out = O / l  (fp32, coalesced across the 16-lane halves)
#pragma unroll
  for (int r = 0; r < 8; ++r) {
    const float inv = 1.0f / lrow[r];
    const size_t row = (size_t)(rowBase + g * 8 + r);
#pragma unroll
    for (int c = 0; c < 8; ++c) {
      out[row * D_OUT + c * 16 + n] = o[c][r] * inv;
    }
  }
}

// ---------------------------------------------------------------------------
extern "C" void kernel_launch(void* const* d_in, const int* in_sizes, int n_in,
                              void* d_out, int out_size, void* d_ws, size_t ws_size,
                              hipStream_t stream) {
  const float* x  = (const float*)d_in[0];
  const float* wq = (const float*)d_in[1];
  const float* wk = (const float*)d_in[2];
  const float* wv = (const float*)d_in[3];
  float* out = (float*)d_out;

  // workspace: Qh (2MB) | Kt (2MB) | Vh (2MB), all f16
  half_t* Qh = (half_t*)d_ws;
  half_t* Kt = Qh + (size_t)SEQ * D_OUT;
  half_t* Vh = Kt + (size_t)D_OUT * SEQ;

  qkv_proj_kernel<<<dim3(SEQ / 16, 6), 128, 0, stream>>>(x, wq, wk, wv, Qh, Kt, Vh);
  flash_attn_kernel<<<SEQ / 16, 32, 0, stream>>>(Qh, Kt, Vh, out);
}
